// MultiheadAttention_12412455486067
// MI455X (gfx1250) — matrix operand order repaired and checked by reading, not compiled, not run on hardware
//
#include <hip/hip_runtime.h>

// ---------------- types & helpers ----------------
typedef __attribute__((ext_vector_type(16))) __bf16 bf16x16;
typedef __attribute__((ext_vector_type(8)))  __bf16 bf16x8;
typedef __attribute__((ext_vector_type(8)))  float  f32x8;
typedef __attribute__((ext_vector_type(4)))  float  f32x4;

#define T_LEN   1024
#define N_B     4
#define D_MODEL 1024
#define H_NUM   16
#define D_HEAD  64
#define TN      4096          // T * N  (rows of the projections)
#define D3      3072          // 3 * D_MODEL

static __device__ __forceinline__ __bf16 f2bf(float f) {
    return (__bf16)f;                         // native v_cvt bf16 path
}
static __device__ __forceinline__ unsigned f2bf2(float lo, float hi) {
    union { __bf16 b[2]; unsigned u; } x;
    x.b[0] = (__bf16)lo; x.b[1] = (__bf16)hi;
    return x.u;
}
static __device__ __forceinline__ bf16x8 ld8(const __bf16* p) {
    return *reinterpret_cast<const bf16x8*>(p);
}
static __device__ __forceinline__ bf16x16 cat16(bf16x8 lo, bf16x8 hi) {
    return __builtin_shufflevector(lo, hi, 0,1,2,3,4,5,6,7,8,9,10,11,12,13,14,15);
}
static __device__ __forceinline__ f32x8 wmma_bf16(bf16x16 a, bf16x16 b, f32x8 c) {
    // D = A(16x32) * B(32x16) + C, f32 accumulate
    return __builtin_amdgcn_wmma_f32_16x16x32_bf16(false, a, false, b, (short)0, c,
                                                   false, false);
}

// gfx1250 async global->LDS copy (16B per lane), tracked by ASYNCcnt.
// Low 32 bits of a generic shared pointer are the LDS byte offset (aperture rule).
static __device__ __forceinline__ void async_ld_b128(void* lds, const void* g) {
    unsigned l = (unsigned)(uintptr_t)lds;
    unsigned long long a = (unsigned long long)(uintptr_t)g;
    asm volatile("global_load_async_to_lds_b128 %0, %1, off"
                 :: "v"(l), "v"(a) : "memory");
}
static __device__ __forceinline__ void wait_async0() {
    asm volatile("s_wait_asynccnt 0" ::: "memory");
}

// ---------------- shared WMMA GEMM mainloop ----------------
// C_tile(128x64) = A(f32, row-major, ld=1024)[m_blk..] * B^T where
// B(f32, row-major, ld=1024) holds one output column per row (NT gemm).
// Block = 256 threads = 8 waves laid out 4(m) x 2(n); each wave owns 32x32.
static __device__ __forceinline__ void gemm_mainloop(
    const float* __restrict__ Ag, const float* __restrict__ Bg,
    int m_blk, int j_blk, int tid,
    __bf16 (&As)[128][40], __bf16 (&Bs)[64][40], f32x8 (&acc)[2][2])
{
    const int lane = tid & 31;
    const int wv   = tid >> 5;
    const int wm   = wv >> 1;           // 0..3
    const int wn   = wv & 1;            // 0..1
    const int hl   = lane >> 4;         // half-wave
    const int ln   = lane & 15;

    const int arow = tid >> 1, ahalf = tid & 1;   // A: 128 rows x 32 k
    const int bcol = tid >> 2, bq    = tid & 3;   // B: 64 cols x 32 k

    for (int kb = 0; kb < 1024; kb += 32) {
        if (kb + 32 < 1024) {  // prefetch next K slab (global_prefetch_b8)
            __builtin_prefetch(Ag + (m_blk + arow) * 1024 + kb + 32 + ahalf * 16, 0, 3);
            __builtin_prefetch(Bg + (j_blk + bcol) * 1024 + kb + 32 + bq * 8, 0, 3);
        }
        { // A tile: 16 floats per thread -> bf16 LDS
            const float* src = Ag + (m_blk + arow) * 1024 + kb + ahalf * 16;
            f32x4 v0 = *(const f32x4*)(src);
            f32x4 v1 = *(const f32x4*)(src + 4);
            f32x4 v2 = *(const f32x4*)(src + 8);
            f32x4 v3 = *(const f32x4*)(src + 12);
            unsigned* dst = (unsigned*)&As[arow][ahalf * 16];
            dst[0] = f2bf2(v0[0], v0[1]); dst[1] = f2bf2(v0[2], v0[3]);
            dst[2] = f2bf2(v1[0], v1[1]); dst[3] = f2bf2(v1[2], v1[3]);
            dst[4] = f2bf2(v2[0], v2[1]); dst[5] = f2bf2(v2[2], v2[3]);
            dst[6] = f2bf2(v3[0], v3[1]); dst[7] = f2bf2(v3[2], v3[3]);
        }
        { // B tile (column-major in LDS == rows of Bg)
            const float* src = Bg + (j_blk + bcol) * 1024 + kb + bq * 8;
            f32x4 v0 = *(const f32x4*)(src);
            f32x4 v1 = *(const f32x4*)(src + 4);
            unsigned* dst = (unsigned*)&Bs[bcol][bq * 8];
            dst[0] = f2bf2(v0[0], v0[1]); dst[1] = f2bf2(v0[2], v0[3]);
            dst[2] = f2bf2(v1[0], v1[1]); dst[3] = f2bf2(v1[2], v1[3]);
        }
        __syncthreads();

        bf16x16 afr[2], bfr[2];
        #pragma unroll
        for (int mi = 0; mi < 2; ++mi) {
            // A frag: lanes<16 -> K{0..7,16..23}; lanes>=16 -> K{8..15,24..31}
            const __bf16* p = &As[wm * 32 + mi * 16 + ln][hl * 8];
            afr[mi] = cat16(ld8(p), ld8(p + 16));
        }
        #pragma unroll
        for (int ni = 0; ni < 2; ++ni) {
            // B frag: lanes<16 -> K{0..7,16..23}; lanes>=16 -> K{8..15,24..31}
            const __bf16* p = &Bs[wn * 32 + ni * 16 + ln][hl * 8];
            bfr[ni] = cat16(ld8(p), ld8(p + 16));
        }
        #pragma unroll
        for (int mi = 0; mi < 2; ++mi)
            #pragma unroll
            for (int ni = 0; ni < 2; ++ni)
                acc[mi][ni] = wmma_bf16(afr[mi], bfr[ni], acc[mi][ni]);
        __syncthreads();
    }
}

// ---------------- kernel 1: QKV projection ----------------
// qkv = x @ in_proj_w^T + b ; scatter into Q[bh][t][d] (pre-scaled by 1/8),
// K[bh][t][d], V transposed [bh][d][t]   (all bf16)
__global__ __launch_bounds__(256) void qkv_kernel(
    const float* __restrict__ x, const float* __restrict__ w,
    const float* __restrict__ bias,
    __bf16* __restrict__ qB, __bf16* __restrict__ kB, __bf16* __restrict__ vT)
{
    __shared__ __bf16 As[128][40];
    __shared__ __bf16 Bs[64][40];

    const int tid  = threadIdx.x;
    const int lane = tid & 31;
    const int wv   = tid >> 5;
    const int wm   = wv >> 1, wn = wv & 1;
    const int hl   = lane >> 4, ln = lane & 15;
    const int m_blk = blockIdx.x * 128;
    const int j_blk = blockIdx.y * 64;
    const int seg   = j_blk >> 10;         // 0=Q, 1=K, 2=V  (uniform per block)

    f32x8 acc[2][2] = {};
    gemm_mainloop(x, w, m_blk, j_blk, tid, As, Bs, acc);

    #pragma unroll
    for (int mi = 0; mi < 2; ++mi)
        #pragma unroll
        for (int ni = 0; ni < 2; ++ni) {
            const int j  = j_blk + wn * 32 + ni * 16 + ln;     // output column
            const int jj = j & 1023;                           // within segment
            const int h = jj >> 6, d = jj & 63;
            const float bv = bias[j];
            const int mbase = m_blk + wm * 32 + mi * 16 + hl * 8;
            if (seg == 0) {                 // Q (scale by 1/sqrt(Dh))
                #pragma unroll
                for (int r = 0; r < 8; ++r) {
                    const int m = mbase + r, t = m >> 2, n = m & 3;
                    qB[((n * 16 + h) * 1024 + t) * 64 + d] =
                        f2bf((acc[mi][ni][r] + bv) * 0.125f);
                }
            } else if (seg == 1) {          // K
                #pragma unroll
                for (int r = 0; r < 8; ++r) {
                    const int m = mbase + r, t = m >> 2, n = m & 3;
                    kB[((n * 16 + h) * 1024 + t) * 64 + d] =
                        f2bf(acc[mi][ni][r] + bv);
                }
            } else {                        // V, transposed [bh][d][t]
                #pragma unroll
                for (int r = 0; r < 8; ++r) {
                    const int m = mbase + r, t = m >> 2, n = m & 3;
                    vT[((n * 16 + h) * 64 + d) * 1024 + t] =
                        f2bf(acc[mi][ni][r] + bv);
                }
            }
        }
}

// ---------------- kernel 2: attention (per bh, per 32-row t tile) ----------
// Pass 1: online row-max m and Z = sum_s exp(l-m)*mult  (softmax denom cancels)
// Pass 2: w = exp(l-m)*mult/Z -> d_out, and attn += w @ V via WMMA.
__global__ __launch_bounds__(64) void attn_kernel(
    const __bf16* __restrict__ qB, const __bf16* __restrict__ kB,
    const __bf16* __restrict__ vT,
    const float* __restrict__ mask, const float* __restrict__ mult,
    float* __restrict__ wOut, float* __restrict__ attnOut)
{
    __shared__ __bf16 Qs[32][72];
    __shared__ __bf16 Ks[32][72];
    __shared__ __bf16 Vs[64][40];   // [d][s]
    __shared__ __bf16 Ws[32][40];   // w tile, bf16 for PV A-fragment

    const int tid  = threadIdx.x;
    const int lane = tid & 31;
    const int wv   = tid >> 5;              // 2 waves: rows 0-15 / 16-31
    const int hl   = lane >> 4, ln = lane & 15;
    const int t0   = blockIdx.x * 32;
    const int bh   = blockIdx.y;            // n*16 + h
    const int n    = bh >> 4, h = bh & 15;

    const int row  = tid >> 1, half = tid & 1;   // tile-fill mapping (32x64)

    { // Q tile via async global->LDS (ASYNCcnt path)
        const __bf16* src = qB + (bh * 1024 + t0 + row) * 64 + half * 32;
        __bf16* dst = &Qs[row][half * 32];
        async_ld_b128(dst +  0, src +  0);
        async_ld_b128(dst +  8, src +  8);
        async_ld_b128(dst + 16, src + 16);
        async_ld_b128(dst + 24, src + 24);
        wait_async0();
    }
    __syncthreads();

    bf16x16 aq[2];                           // Q fragments (two 32-wide K steps)
    #pragma unroll
    for (int kk = 0; kk < 2; ++kk) {
        const __bf16* p = &Qs[wv * 16 + ln][kk * 32 + hl * 8];
        aq[kk] = cat16(ld8(p), ld8(p + 16));
    }

    float mrow[8], zrow[8];
    #pragma unroll
    for (int r = 0; r < 8; ++r) { mrow[r] = -3.4e38f; zrow[r] = 0.0f; }

    const int tRow_base = t0 + wv * 16 + hl * 8;

    // ---------------- pass 1: stats ----------------
    for (int st = 0; st < 32; ++st) {
        const int s0 = st * 32;
        { const __bf16* src = kB + (bh * 1024 + s0 + row) * 64 + half * 32;
          __bf16* dst = &Ks[row][half * 32];
          async_ld_b128(dst +  0, src +  0);
          async_ld_b128(dst +  8, src +  8);
          async_ld_b128(dst + 16, src + 16);
          async_ld_b128(dst + 24, src + 24);
          wait_async0(); }
        __syncthreads();

        f32x8 c[2] = {};
        #pragma unroll
        for (int ni = 0; ni < 2; ++ni)
            #pragma unroll
            for (int kk = 0; kk < 2; ++kk) {
                const __bf16* p = &Ks[ni * 16 + ln][kk * 32 + hl * 8];
                c[ni] = wmma_bf16(aq[kk], cat16(ld8(p), ld8(p + 16)), c[ni]);
            }

        const int sc0 = s0 + ln, sc1 = s0 + 16 + ln;
        float v0[8], v1[8];
        #pragma unroll
        for (int r = 0; r < 8; ++r) {
            const int tr = tRow_base + r;
            v0[r] = c[0][r] + mask[tr * 1024 + sc0];
            v1[r] = c[1][r] + mask[tr * 1024 + sc1];
        }
        #pragma unroll
        for (int r = 0; r < 8; ++r) {
            float tm = fmaxf(v0[r], v1[r]);
            for (int off = 1; off < 16; off <<= 1)
                tm = fmaxf(tm, __shfl_xor(tm, off));
            const float mn = fmaxf(mrow[r], tm);
            const int tr = tRow_base + r;
            float p = __expf(v0[r] - mn) * mult[(tr * 1024 + sc0) * 4 + n]
                    + __expf(v1[r] - mn) * mult[(tr * 1024 + sc1) * 4 + n];
            for (int off = 1; off < 16; off <<= 1)
                p += __shfl_xor(p, off);
            zrow[r] = zrow[r] * __expf(mrow[r] - mn) + p;
            mrow[r] = mn;
        }
        __syncthreads();
    }

    float rz[8];
    #pragma unroll
    for (int r = 0; r < 8; ++r) rz[r] = 1.0f / zrow[r];

    // ---------------- pass 2: w out + attn = w @ V ----------------
    f32x8 accD[4] = {};
    for (int st = 0; st < 32; ++st) {
        const int s0 = st * 32;
        { const __bf16* src = kB + (bh * 1024 + s0 + row) * 64 + half * 32;
          __bf16* dst = &Ks[row][half * 32];
          async_ld_b128(dst +  0, src +  0);
          async_ld_b128(dst +  8, src +  8);
          async_ld_b128(dst + 16, src + 16);
          async_ld_b128(dst + 24, src + 24); }
        { // V tile, already transposed in workspace: contiguous in s
          const __bf16* src = vT + (bh * 64 + tid) * 1024 + s0;
          __bf16* dst = &Vs[tid][0];
          async_ld_b128(dst +  0, src +  0);
          async_ld_b128(dst +  8, src +  8);
          async_ld_b128(dst + 16, src + 16);
          async_ld_b128(dst + 24, src + 24); }
        wait_async0();
        __syncthreads();

        f32x8 c[2] = {};
        #pragma unroll
        for (int ni = 0; ni < 2; ++ni)
            #pragma unroll
            for (int kk = 0; kk < 2; ++kk) {
                const __bf16* p = &Ks[ni * 16 + ln][kk * 32 + hl * 8];
                c[ni] = wmma_bf16(aq[kk], cat16(ld8(p), ld8(p + 16)), c[ni]);
            }

        #pragma unroll
        for (int ni = 0; ni < 2; ++ni) {
            const int sc = s0 + ni * 16 + ln;
            #pragma unroll
            for (int r = 0; r < 8; ++r) {
                const int tr = tRow_base + r;
                const float val = c[ni][r] + mask[tr * 1024 + sc];
                const float wval = __expf(val - mrow[r]) *
                                   mult[(tr * 1024 + sc) * 4 + n] * rz[r];
                wOut[tr * 65536 + sc * 64 + bh] = wval;              // (T,S,N,H)
                Ws[wv * 16 + hl * 8 + r][ni * 16 + ln] = f2bf(wval);
            }
        }
        // PV: A = w (16x32, K=s), B = V^T fragments (cols = d)
        bf16x16 aw;
        { const __bf16* p = &Ws[wv * 16 + ln][hl * 8];
          aw = cat16(ld8(p), ld8(p + 16)); }
        #pragma unroll
        for (int nd = 0; nd < 4; ++nd) {
            const __bf16* p = &Vs[nd * 16 + ln][hl * 8];
            accD[nd] = wmma_bf16(aw, cat16(ld8(p), ld8(p + 16)), accD[nd]);
        }
        __syncthreads();
    }

    #pragma unroll
    for (int nd = 0; nd < 4; ++nd) {
        const int d = nd * 16 + ln;
        #pragma unroll
        for (int r = 0; r < 8; ++r) {
            const int tr = tRow_base + r;
            attnOut[(tr * 4 + n) * 1024 + h * 64 + d] = accD[nd][r];
        }
    }
}

// ---------------- kernel 3: output projection ----------------
__global__ __launch_bounds__(256) void outproj_kernel(
    const float* __restrict__ attn, const float* __restrict__ w,
    const float* __restrict__ bias, float* __restrict__ out)
{
    __shared__ __bf16 As[128][40];
    __shared__ __bf16 Bs[64][40];

    const int tid  = threadIdx.x;
    const int lane = tid & 31;
    const int wv   = tid >> 5;
    const int wm   = wv >> 1, wn = wv & 1;
    const int hl   = lane >> 4, ln = lane & 15;
    const int m_blk = blockIdx.x * 128;
    const int j_blk = blockIdx.y * 64;

    f32x8 acc[2][2] = {};
    gemm_mainloop(attn, w, m_blk, j_blk, tid, As, Bs, acc);

    #pragma unroll
    for (int mi = 0; mi < 2; ++mi)
        #pragma unroll
        for (int ni = 0; ni < 2; ++ni) {
            const int j = j_blk + wn * 32 + ni * 16 + ln;
            const float bv = bias[j];
            const int mbase = m_blk + wm * 32 + mi * 16 + hl * 8;
            #pragma unroll
            for (int r = 0; r < 8; ++r)
                out[(mbase + r) * 1024 + j] = acc[mi][ni][r] + bv;
        }
}

// ---------------- host launch ----------------
extern "C" void kernel_launch(void* const* d_in, const int* in_sizes, int n_in,
                              void* d_out, int out_size, void* d_ws, size_t ws_size,
                              hipStream_t stream) {
    const float* x    = (const float*)d_in[0];
    const float* mask = (const float*)d_in[1];
    const float* mult = (const float*)d_in[2];
    const float* ipw  = (const float*)d_in[3];
    const float* ipb  = (const float*)d_in[4];
    const float* opw  = (const float*)d_in[5];
    const float* opb  = (const float*)d_in[6];

    float* out  = (float*)d_out;                 // (T, N, D)
    float* wOut = out + (size_t)TN * D_MODEL;    // (T, S, N, H)

    const size_t QKV_ELEMS = (size_t)N_B * H_NUM * T_LEN * D_HEAD;  // 4,194,304
    char* ws = (char*)d_ws;
    __bf16* qB   = (__bf16*)ws;
    __bf16* kB   = qB + QKV_ELEMS;
    __bf16* vT   = kB + QKV_ELEMS;
    float*  attn = (float*)(ws + 3 * QKV_ELEMS * sizeof(__bf16));

    qkv_kernel<<<dim3(TN / 128, D3 / 64), 256, 0, stream>>>(x, ipw, ipb, qB, kB, vT);
    attn_kernel<<<dim3(T_LEN / 32, N_B * H_NUM), 64, 0, stream>>>(
        qB, kB, vT, mask, mult, wOut, attn);
    outproj_kernel<<<dim3(TN / 128, D_MODEL / 64), 256, 0, stream>>>(
        attn, opw, opb, out);
}
